// Head_24799141167224
// MI455X (gfx1250) — matrix operand order repaired and checked by reading, not compiled, not run on hardware
//
#include <hip/hip_runtime.h>

typedef __attribute__((ext_vector_type(16))) _Float16 v16h;
typedef __attribute__((ext_vector_type(4)))  _Float16 v4h;
typedef __attribute__((ext_vector_type(8)))  float    v8f;

#define NE      2048
#define DHEAD   128
#define NQKV    384
#define BATCH   4
#define TLEN    4096
#define TOKENS  (BATCH * TLEN)     // 16384
#define KTOP    409
#define KTP     416                // 26 * 16 (padded)
#define JTILES  26

// ---------------------------------------------------------------------------
// helpers: load a 16-half fragment from two 16B chunks
// A-fragment (16x32 f16, row-major source): lane&15 = row, lane>=16 -> k+16
// halves 0..15 contiguous at base
__device__ __forceinline__ v16h load_frag_a(const _Float16* base) {
    v16h a;
    const uint4* p = reinterpret_cast<const uint4*>(base);
    reinterpret_cast<uint4*>(&a)[0] = p[0];
    reinterpret_cast<uint4*>(&a)[1] = p[1];   // +8 halves
    return a;
}
// B-fragment (32x16 f16, K-contiguous per column): lane&15 = col,
// lane>=16 -> k+16; halves 0..15 contiguous at base
__device__ __forceinline__ v16h load_frag_b(const _Float16* base) {
    v16h b;
    const uint4* p = reinterpret_cast<const uint4*>(base);
    reinterpret_cast<uint4*>(&b)[0] = p[0];
    reinterpret_cast<uint4*>(&b)[1] = p[1];   // +8 halves
    return b;
}

__device__ __forceinline__ v4h cvt4(const float4 f) {
    v4h h;
    h.x = (_Float16)f.x; h.y = (_Float16)f.y;
    h.z = (_Float16)f.z; h.w = (_Float16)f.w;
    return h;
}

// ---------------------------------------------------------------------------
// 0) build Wt f16 [384, 2048]: rows 0..127 = Wq^T, 128..255 = Wk^T, 256..383 = Wv^T
__global__ void build_wt(const float* __restrict__ Wq, const float* __restrict__ Wk,
                         const float* __restrict__ Wv, _Float16* __restrict__ Wt) {
    int gid = blockIdx.x * 256 + threadIdx.x;
    if (gid >= NQKV * NE) return;
    int k = gid & (NE - 1);
    int n = gid >> 11;
    const float* W = (n < 128) ? Wq : ((n < 256) ? Wk : Wv);
    int c = n & 127;
    Wt[(size_t)n * NE + k] = (_Float16)W[(size_t)k * DHEAD + c];
}

__global__ void zerof(float* __restrict__ p, int n) {
    int i = blockIdx.x * 256 + threadIdx.x;
    if (i < n) p[i] = 0.0f;
}

// ---------------------------------------------------------------------------
// 1) QKV GEMM: D[16384, 384] = index[16384, 2048] * [Wq|Wk|Wv]
// block = 256 threads (8 waves), 32-row M-tile per block (2 A-frags).
// A tile (32x32) staged f32->f16 via double-buffered LDS: all 256 threads
// move one float4 each (branch-free). B fragments software-pipelined one
// K-step ahead in registers. Each wave: 2 A-frags x 3 N-tiles = 6 WMMA/K-step.
__global__ void gemm_qkv(const float* __restrict__ A, const _Float16* __restrict__ Wt,
                         float* __restrict__ D) {
    __shared__ __align__(16) _Float16 As[2][32 * 32];
    const int m0   = blockIdx.x * 32;
    const int wave = threadIdx.x >> 5;
    const int lane = threadIdx.x & 31;
    const int row  = lane & 15;
    const int hi   = lane >> 4;          // 0 or 1
    const int n0   = wave * 48;

    // staging: thread t moves float4 at (row t>>3, col (t&7)*4) of the 32x32 tile
    const int  srow = threadIdx.x >> 3;        // 0..31
    const int  scol = (threadIdx.x & 7) * 4;   // 0,4,...,28
    const float* aptr = A + (size_t)(m0 + srow) * NE + scol;

    v8f acc00 = {}, acc01 = {}, acc02 = {};   // A rows 0..15  x 3 N-tiles
    v8f acc10 = {}, acc11 = {}, acc12 = {};   // A rows 16..31 x 3 N-tiles

    // prologue: stage tile 0
    {
        float4 f = *reinterpret_cast<const float4*>(aptr);
        *reinterpret_cast<v4h*>(&As[0][srow * 32 + scol]) = cvt4(f);
    }
    __syncthreads();

    const _Float16* wbase = Wt + (size_t)(n0 + row) * NE + (hi ? 16 : 0);
    // B fragments for iteration 0
    v16h b0 = load_frag_b(wbase);
    v16h b1 = load_frag_b(wbase + (size_t)16 * NE);
    v16h b2 = load_frag_b(wbase + (size_t)32 * NE);

    const int nIter = NE / 32;   // 64
    for (int i = 0; i < nIter; ++i) {
        const int cur   = i & 1;
        const int inext = (i + 1 < nIter) ? (i + 1) : 0;   // clamp: dead work on last iter

        // prefetch next A tile (global) and next B fragments (global) early
        float4 fn = *reinterpret_cast<const float4*>(aptr + (size_t)inext * 32);
        const _Float16* wb = wbase + inext * 32;
        v16h nb0 = load_frag_b(wb);
        v16h nb1 = load_frag_b(wb + (size_t)16 * NE);
        v16h nb2 = load_frag_b(wb + (size_t)32 * NE);

        // current A fragments from LDS
        v16h a0 = load_frag_a(&As[cur][row * 32 + (hi ? 16 : 0)]);
        v16h a1 = load_frag_a(&As[cur][(16 + row) * 32 + (hi ? 16 : 0)]);

        acc00 = __builtin_amdgcn_wmma_f32_16x16x32_f16(false, a0, false, b0, (short)0, acc00, false, false);
        acc01 = __builtin_amdgcn_wmma_f32_16x16x32_f16(false, a0, false, b1, (short)0, acc01, false, false);
        acc02 = __builtin_amdgcn_wmma_f32_16x16x32_f16(false, a0, false, b2, (short)0, acc02, false, false);
        acc10 = __builtin_amdgcn_wmma_f32_16x16x32_f16(false, a1, false, b0, (short)0, acc10, false, false);
        acc11 = __builtin_amdgcn_wmma_f32_16x16x32_f16(false, a1, false, b1, (short)0, acc11, false, false);
        acc12 = __builtin_amdgcn_wmma_f32_16x16x32_f16(false, a1, false, b2, (short)0, acc12, false, false);

        // stage next A tile into the other LDS buffer (dead store on last iter)
        *reinterpret_cast<v4h*>(&As[cur ^ 1][srow * 32 + scol]) = cvt4(fn);
        __syncthreads();

        b0 = nb0; b1 = nb1; b2 = nb2;
    }

    const int rbase = m0 + (hi ? 8 : 0);
    #pragma unroll
    for (int i = 0; i < 8; ++i) {
        float* drow0 = D + (size_t)(rbase + i) * NQKV;
        drow0[n0 + row +  0] = acc00[i];
        drow0[n0 + row + 16] = acc01[i];
        drow0[n0 + row + 32] = acc02[i];
        float* drow1 = D + (size_t)(rbase + 16 + i) * NQKV;
        drow1[n0 + row +  0] = acc10[i];
        drow1[n0 + row + 16] = acc11[i];
        drow1[n0 + row + 32] = acc12[i];
    }
}

// ---------------------------------------------------------------------------
// 2) q norms: one wave per row (cols 0..127 of D are q)
__global__ void qnorms(const float* __restrict__ D, float* __restrict__ norms) {
    int row  = blockIdx.x * 8 + (threadIdx.x >> 5);
    int lane = threadIdx.x & 31;
    float s = 0.0f;
    #pragma unroll
    for (int d = lane; d < DHEAD; d += 32) {
        float q = D[(size_t)row * NQKV + d];
        s += q * q;
    }
    #pragma unroll
    for (int off = 16; off; off >>= 1) s += __shfl_down(s, off, 32);
    if (lane == 0) norms[row] = sqrtf(s);
}

// ---------------------------------------------------------------------------
// 3) top-k per batch: bitonic sort 4096 keys (desc by norm, asc by index on ties)
__global__ void topk_kernel(const float* __restrict__ norms, int* __restrict__ topidx) {
    __shared__ unsigned long long keys[TLEN];
    const int b = blockIdx.x;
    const int t = threadIdx.x;             // 1024 threads
    for (int i = t; i < TLEN; i += 1024) {
        unsigned int bits = __float_as_uint(norms[b * TLEN + i]);  // norms >= 0
        keys[i] = ((unsigned long long)bits << 32) | (unsigned int)(TLEN - 1 - i);
    }
    __syncthreads();
    for (int k = 2; k <= TLEN; k <<= 1) {
        for (int j = k >> 1; j > 0; j >>= 1) {
            for (int i = t; i < TLEN; i += 1024) {
                int ixj = i ^ j;
                if (ixj > i) {
                    bool desc = ((i & k) == 0);
                    unsigned long long a = keys[i], c = keys[ixj];
                    bool sw = desc ? (a < c) : (a > c);
                    if (sw) { keys[i] = c; keys[ixj] = a; }
                }
            }
            __syncthreads();
        }
    }
    for (int i = t; i < KTOP; i += 1024)
        topidx[b * KTOP + i] = (TLEN - 1) - (int)(keys[i] & 0xFFFFFFFFull);
}

// ---------------------------------------------------------------------------
// 4) gather top-k rows -> f16 qh/kh [B,416,128], vhT [B,128,416]; pad rows = 0
__global__ void gather_topk(const float* __restrict__ D, const int* __restrict__ topidx,
                            _Float16* __restrict__ qh, _Float16* __restrict__ kh,
                            _Float16* __restrict__ vhT) {
    int gid = blockIdx.x * 256 + threadIdx.x;     // B*KTP*128 total
    if (gid >= BATCH * KTP * DHEAD) return;
    int d = gid & 127;
    int j = (gid >> 7) % KTP;
    int b = gid / (KTP * DHEAD);
    _Float16 qv = (_Float16)0.0f, kv = (_Float16)0.0f, vv = (_Float16)0.0f;
    if (j < KTOP) {
        int tok = topidx[b * KTOP + j];
        const float* p = D + (size_t)(b * TLEN + tok) * NQKV;
        qv = (_Float16)p[d];
        kv = (_Float16)p[DHEAD + d];
        vv = (_Float16)p[2 * DHEAD + d];
    }
    qh [((size_t)b * KTP + j) * DHEAD + d] = qv;
    kh [((size_t)b * KTP + j) * DHEAD + d] = kv;
    vhT[((size_t)b * DHEAD + d) * KTP + j] = vv;
}

// ---------------------------------------------------------------------------
// 5) scores: S[b, 416, 416] = (qh * kh^T) / sqrt(128); one wave per 16x16 tile
__global__ void scores_kernel(const _Float16* __restrict__ qh, const _Float16* __restrict__ kh,
                              float* __restrict__ S) {
    int wid = blockIdx.x * (blockDim.x >> 5) + (threadIdx.x >> 5);
    if (wid >= BATCH * JTILES * JTILES) return;
    int jt = wid % JTILES;
    int it = (wid / JTILES) % JTILES;
    int b  = wid / (JTILES * JTILES);
    int lane = threadIdx.x & 31;
    int rr = lane & 15, hi = lane >> 4;

    v8f acc = {};
    const _Float16* qb = qh + ((size_t)b * KTP + it * 16 + rr) * DHEAD;
    const _Float16* kb = kh + ((size_t)b * KTP + jt * 16 + rr) * DHEAD;
    #pragma unroll
    for (int kt = 0; kt < DHEAD; kt += 32) {
        v16h a  = load_frag_a(qb + kt + (hi ? 16 : 0));
        v16h bb = load_frag_b(kb + kt + (hi ? 16 : 0));
        acc = __builtin_amdgcn_wmma_f32_16x16x32_f16(false, a, false, bb, (short)0, acc, false, false);
    }
    const float scale = 0.08838834764831845f;   // 1/sqrt(128)
    int i0 = it * 16 + (hi ? 8 : 0);
    int j  = jt * 16 + rr;
    #pragma unroll
    for (int i = 0; i < 8; ++i)
        S[((size_t)b * KTP + i0 + i) * KTP + j] = acc[i] * scale;
}

// ---------------------------------------------------------------------------
// 6) stable softmax over valid cols (j < 409); write f16 probs, pad cols = 0
__global__ void softmax_kernel(const float* __restrict__ S, _Float16* __restrict__ wh) {
    const int rowg = blockIdx.x;                 // b*KTP + i
    const float* srow = S + (size_t)rowg * KTP;
    __shared__ float redm[8];
    __shared__ float reds[8];
    int t = threadIdx.x, lane = t & 31, w = t >> 5;

    float m = -1e30f;
    for (int j = t; j < KTOP; j += 256) m = fmaxf(m, srow[j]);
    #pragma unroll
    for (int off = 16; off; off >>= 1) m = fmaxf(m, __shfl_down(m, off, 32));
    if (lane == 0) redm[w] = m;
    __syncthreads();
    float mm = redm[0];
    #pragma unroll
    for (int i = 1; i < 8; ++i) mm = fmaxf(mm, redm[i]);

    float s = 0.0f;
    for (int j = t; j < KTOP; j += 256) s += __expf(srow[j] - mm);
    #pragma unroll
    for (int off = 16; off; off >>= 1) s += __shfl_down(s, off, 32);
    if (lane == 0) reds[w] = s;
    __syncthreads();
    float ss = 0.0f;
    #pragma unroll
    for (int i = 0; i < 8; ++i) ss += reds[i];
    float inv = 1.0f / ss;

    _Float16* wrow = wh + (size_t)rowg * KTP;
    for (int j = t; j < KTP; j += 256) {
        float val = (j < KTOP) ? __expf(srow[j] - mm) * inv : 0.0f;
        wrow[j] = (_Float16)val;
    }
}

// ---------------------------------------------------------------------------
// 7) out = wh @ v_topk, scattered to d_out rows topidx.  8 waves/block = all of D.
__global__ void pv_scatter(const _Float16* __restrict__ wh, const _Float16* __restrict__ vhT,
                           const int* __restrict__ topidx, float* __restrict__ out) {
    const int it = blockIdx.x % JTILES;
    const int b  = blockIdx.x / JTILES;
    const int w  = threadIdx.x >> 5;
    const int lane = threadIdx.x & 31;
    const int rr = lane & 15, hi = lane >> 4;
    const int n0 = w * 16;

    v8f acc = {};
    const _Float16* ab = wh  + ((size_t)b * KTP + it * 16 + rr) * KTP;
    const _Float16* bb = vhT + ((size_t)b * DHEAD + n0 + rr) * KTP;
    #pragma unroll
    for (int kt = 0; kt < KTP; kt += 32) {
        v16h a = load_frag_a(ab + kt + (hi ? 16 : 0));
        v16h v = load_frag_b(bb + kt + (hi ? 16 : 0));
        acc = __builtin_amdgcn_wmma_f32_16x16x32_f16(false, a, false, v, (short)0, acc, false, false);
    }
    int i0 = it * 16 + (hi ? 8 : 0);
    #pragma unroll
    for (int i = 0; i < 8; ++i) {
        int irow = i0 + i;
        if (irow < KTOP) {
            int tok = topidx[b * KTOP + irow];
            out[((size_t)(b * TLEN + tok)) * DHEAD + n0 + rr] = acc[i];
        }
    }
}

// ---------------------------------------------------------------------------
extern "C" void kernel_launch(void* const* d_in, const int* in_sizes, int n_in,
                              void* d_out, int out_size, void* d_ws, size_t ws_size,
                              hipStream_t stream) {
    const float* index = (const float*)d_in[0];
    const float* Wq    = (const float*)d_in[1];
    const float* Wk    = (const float*)d_in[2];
    const float* Wv    = (const float*)d_in[3];
    float* out = (float*)d_out;

    // workspace carve-up (256B aligned)
    char* ws = (char*)d_ws;
    size_t off = 0;
    auto alloc = [&](size_t bytes) { void* p = ws + off; off = (off + bytes + 255) & ~(size_t)255; return p; };
    _Float16* Wt    = (_Float16*)alloc((size_t)NQKV * NE * 2);          // 1.5 MB
    float*    Dqkv  = (float*)   alloc((size_t)TOKENS * NQKV * 4);      // 24 MB
    float*    norms = (float*)   alloc((size_t)TOKENS * 4);
    int*      tidx  = (int*)     alloc((size_t)BATCH * KTOP * 4);
    _Float16* qh    = (_Float16*)alloc((size_t)BATCH * KTP * DHEAD * 2);
    _Float16* kh    = (_Float16*)alloc((size_t)BATCH * KTP * DHEAD * 2);
    _Float16* vhT   = (_Float16*)alloc((size_t)BATCH * DHEAD * KTP * 2);
    float*    Sbuf  = (float*)   alloc((size_t)BATCH * KTP * KTP * 4);  // 2.8 MB
    _Float16* wh    = (_Float16*)alloc((size_t)BATCH * KTP * KTP * 2);  // 1.4 MB
    (void)ws_size; (void)in_sizes; (void)n_in; (void)out_size;

    build_wt<<<(NQKV * NE + 255) / 256, 256, 0, stream>>>(Wq, Wk, Wv, Wt);
    zerof<<<(TOKENS * DHEAD) / 256, 256, 0, stream>>>(out, TOKENS * DHEAD);

    gemm_qkv<<<TOKENS / 32, 256, 0, stream>>>(index, Wt, Dqkv);
    qnorms<<<TOKENS / 8, 256, 0, stream>>>(Dqkv, norms);
    topk_kernel<<<BATCH, 1024, 0, stream>>>(norms, tidx);
    gather_topk<<<(BATCH * KTP * DHEAD + 255) / 256, 256, 0, stream>>>(Dqkv, tidx, qh, kh, vhT);
    scores_kernel<<<(BATCH * JTILES * JTILES + 7) / 8, 256, 0, stream>>>(qh, kh, Sbuf);
    softmax_kernel<<<BATCH * KTP, 256, 0, stream>>>(Sbuf, wh);
    pv_scatter<<<BATCH * JTILES, 256, 0, stream>>>(wh, vhT, tidx, out);
}
